// Attention_32323923869749
// MI455X (gfx1250) — matrix operand order repaired and checked by reading, not compiled, not run on hardware
//
#include <hip/hip_runtime.h>
#include <hip/hip_bf16.h>

// ---------------------------------------------------------------------------
// CDNA5 (gfx1250) fused attention:
//   q=query@Wq^T+bq ; k=key@Wk^T+bk ; v=value@Wv^T+bv      (bf16 outputs)
//   S = q k^T ; instance-norm(row) -> softmax == softmax(c*x), c=rsqrt(var)/16
//   O = softmax(c*S) v ;  out = (query + O) @ Wm^T + bm     (fp32 output)
// Matmuls on v_wmma_f32_16x16x32_bf16; K-tile staging via TDM
// (tensor_load_to_lds + s_wait_tensorcnt). Wave32, 8 waves / 256-thread block.
// ---------------------------------------------------------------------------

typedef __attribute__((ext_vector_type(16))) __bf16 v16bf;
typedef __attribute__((ext_vector_type(8)))  float  v8f;

union Frag {            // one 16x32 (A) or 32x16 (B) bf16 fragment per lane
    v16bf v;
    uint4 q[2];
};

#define C_DIM 256
#define N_SEQ 4096
#define LOG2E 1.4426950408889634f

// ---- Tensor Data Mover support --------------------------------------------
#ifndef USE_TDM
#if defined(__has_builtin)
#if __has_builtin(__builtin_amdgcn_tensor_load_to_lds)
#define USE_TDM 1
#endif
#endif
#endif
#ifndef USE_TDM
#define USE_TDM 0
#endif

#if USE_TDM
typedef unsigned int u32x4 __attribute__((ext_vector_type(4)));
typedef int          i32x8 __attribute__((ext_vector_type(8)));
typedef int          i32x4 __attribute__((ext_vector_type(4)));
typedef __attribute__((address_space(3))) unsigned char lds_u8;

__device__ __forceinline__ unsigned lds_addr_of(void* p) {
    return (unsigned)(unsigned long long)(lds_u8*)p;   // generic -> LDS offset
}

// 2D tile DMA: global (row-major, rowDwords 4-byte elems per row) -> LDS.
// D# group0/group1 bitfields per CDNA5 ISA.
// 6-arg builtin form: (g0, g1, g2, g3, g4, cpol); groups 2-4 unused for 2D.
__device__ __forceinline__ void tdm_load_tile_2d(unsigned ldsByteOff,
                                                 const void* gsrc,
                                                 unsigned rowDwords,
                                                 unsigned tileRows,
                                                 unsigned tensorRows)
{
    unsigned long long ga = (unsigned long long)(uintptr_t)gsrc;
    u32x4 g0;
    g0.x = 1u;                                          // count = 1 (valid D#)
    g0.y = ldsByteOff;                                  // lds_addr
    g0.z = (unsigned)ga;                                // global_addr[31:0]
    g0.w = (unsigned)((ga >> 32) & 0x01FFFFFFu) | 0x80000000u; // addr[56:32]|type=2
    i32x8 g1;
    g1[0] = (int)(2u << 16);                            // data_size = 4 bytes
    g1[1] = (int)((rowDwords & 0xFFFFu) << 16);         // tensor_dim0 lo16
    g1[2] = (int)((rowDwords >> 16) | ((tensorRows & 0xFFFFu) << 16));
    g1[3] = (int)((tensorRows >> 16) | (rowDwords << 16)); // dim1 hi | tile_dim0
    g1[4] = (int)(tileRows & 0xFFFFu);                  // tile_dim1 (tile_dim2=0)
    g1[5] = (int)rowDwords;                             // tensor_dim0_stride lo32
    g1[6] = 0;
    g1[7] = 0;
    i32x4 z4 = {0, 0, 0, 0};
    i32x8 z8 = {0, 0, 0, 0, 0, 0, 0, 0};
    __builtin_amdgcn_tensor_load_to_lds(g0, g1, z4, z4, z8, 0);
}
#endif // USE_TDM

__device__ __forceinline__ unsigned short f2bf(float f) {
    unsigned int u = __builtin_bit_cast(unsigned int, f);
    unsigned int r = (u + 0x7fffu + ((u >> 16) & 1u)) >> 16;  // RNE truncate
    return (unsigned short)r;
}

__device__ __forceinline__ v8f wmma_bf16(const Frag& a, const Frag& b, v8f c) {
    return __builtin_amdgcn_wmma_f32_16x16x32_bf16(
        false, a.v, false, b.v, (short)0, c, false, false);
}

// ---------------------------------------------------------------------------
// Generic 256-deep GEMM:  Y[M,256] = (X (+R)) @ W^T + bias
// Block = 128 rows x 64 cols, 8 waves (wave w -> rows 16w..16w+15, all 64 cols)
// fp32 -> bf16 conversion happens during staging, so this stays on VALU loads.
// ---------------------------------------------------------------------------
__global__ void __launch_bounds__(256)
gemm256_kernel(const float* __restrict__ X, const float* __restrict__ R,
               const float* __restrict__ W, const float* __restrict__ bias,
               unsigned short* __restrict__ Ybf, float* __restrict__ Yf, int M)
{
    __shared__ unsigned short ldsX[128 * 32];   // [row][k]  8 KB
    __shared__ unsigned short ldsW[64 * 32];    // [n][k]    4 KB

    const int tid  = threadIdx.x;
    const int lane = tid & 31;
    const int wave = tid >> 5;
    const int half = (lane >> 4) & 1;
    const int l16  = lane & 15;
    const int rowBlk = blockIdx.x * 128;
    const int colBlk = blockIdx.y * 64;

    v8f acc[4] = {};

    for (int ks = 0; ks < 8; ++ks) {
        const int k0 = ks * 32;
        __syncthreads();
        #pragma unroll
        for (int i = 0; i < 4; ++i) {
            int idx = tid + i * 256;
            int r = idx >> 3;
            int c4 = (idx & 7) * 4;
            float4 v = *(const float4*)(X + (size_t)(rowBlk + r) * C_DIM + k0 + c4);
            if (R) {
                float4 rv = *(const float4*)(R + (size_t)(rowBlk + r) * C_DIM + k0 + c4);
                v.x += rv.x; v.y += rv.y; v.z += rv.z; v.w += rv.w;
            }
            ushort4 s; s.x = f2bf(v.x); s.y = f2bf(v.y); s.z = f2bf(v.z); s.w = f2bf(v.w);
            *(ushort4*)(&ldsX[r * 32 + c4]) = s;
        }
        #pragma unroll
        for (int i = 0; i < 2; ++i) {
            int idx = tid + i * 256;
            int n = idx >> 3;
            int c4 = (idx & 7) * 4;
            float4 v = *(const float4*)(W + (size_t)(colBlk + n) * C_DIM + k0 + c4);
            ushort4 s; s.x = f2bf(v.x); s.y = f2bf(v.y); s.z = f2bf(v.z); s.w = f2bf(v.w);
            *(ushort4*)(&ldsW[n * 32 + c4]) = s;
        }
        __syncthreads();

        Frag a;
        const unsigned short* xr = &ldsX[(wave * 16 + l16) * 32];
        a.q[0] = *(const uint4*)(xr + 16 * half);
        a.q[1] = *(const uint4*)(xr + 16 * half + 8);

        #pragma unroll
        for (int nt = 0; nt < 4; ++nt) {
            Frag b;
            const unsigned short* wr = &ldsW[(nt * 16 + l16) * 32 + 16 * half];
            b.q[0] = *(const uint4*)(wr);
            b.q[1] = *(const uint4*)(wr + 8);
            acc[nt] = wmma_bf16(a, b, acc[nt]);
        }
    }

    #pragma unroll
    for (int nt = 0; nt < 4; ++nt) {
        int n = colBlk + nt * 16 + l16;
        float bv = bias[n];
        #pragma unroll
        for (int r = 0; r < 8; ++r) {
            int m = rowBlk + wave * 16 + r + 8 * half;
            float y = acc[nt][r] + bv;
            if (Ybf) Ybf[(size_t)m * C_DIM + n] = f2bf(y);
            if (Yf)  Yf [(size_t)m * C_DIM + n] = y;
        }
    }
}

// ---------------------------------------------------------------------------
// Pass 1: per-row sum / sumsq / max of S = q k^T  (streamed S^T tiles)
// K tile (16x256 bf16 = 16x128 dwords) staged by the Tensor Data Mover.
// ---------------------------------------------------------------------------
__global__ void __launch_bounds__(256)
attn_stats_kernel(const unsigned short* __restrict__ Qb,
                  const unsigned short* __restrict__ Kb,
                  float* __restrict__ cRow, float* __restrict__ cMax)
{
    __shared__ unsigned short ldsK[16 * C_DIM];   // 8 KB

    const int tid  = threadIdx.x;
    const int lane = tid & 31;
    const int wave = tid >> 5;
    const int half = (lane >> 4) & 1;
    const int l16  = lane & 15;
    const int b    = blockIdx.x >> 5;            // 32 q-blocks per batch
    const int qblk = (blockIdx.x & 31) * 128;

    const unsigned short* Qrow = Qb + ((size_t)b * N_SEQ + qblk + wave * 16 + l16) * C_DIM;
    const unsigned short* Kbase = Kb + (size_t)b * N_SEQ * C_DIM;
#if USE_TDM
    const unsigned ldsKoff = lds_addr_of(ldsK);
#endif

    Frag qf[8];
    #pragma unroll
    for (int s = 0; s < 8; ++s) {
        qf[s].q[0] = *(const uint4*)(Qrow + s * 32 + 8 * half);
        qf[s].q[1] = *(const uint4*)(Qrow + s * 32 + 16 + 8 * half);
    }

    float sum = 0.f, ssq = 0.f, mx = -3.0e38f;

    for (int mb = 0; mb < N_SEQ; mb += 16) {
        __syncthreads();
#if USE_TDM
        if (wave == 0) {
            tdm_load_tile_2d(ldsKoff, Kbase + (size_t)mb * C_DIM,
                             C_DIM / 2, 16, N_SEQ - mb);
            __builtin_amdgcn_s_wait_tensorcnt(0);
        }
#else
        #pragma unroll
        for (int i = 0; i < 2; ++i) {
            int idx = tid + i * 256;
            int r  = idx >> 5;
            int c8 = (idx & 31) * 8;
            *(uint4*)(&ldsK[r * C_DIM + c8]) =
                *(const uint4*)(Kbase + (size_t)(mb + r) * C_DIM + c8);
        }
#endif
        __syncthreads();

        v8f st = {};                 // S^T tile: VGPR r -> m=r+8h, lane -> q
        #pragma unroll
        for (int s = 0; s < 8; ++s) {
            Frag a;
            const unsigned short* kr = &ldsK[l16 * C_DIM + s * 32];
            a.q[0] = *(const uint4*)(kr + 8 * half);
            a.q[1] = *(const uint4*)(kr + 16 + 8 * half);
            st = wmma_bf16(a, qf[s], st);
        }
        #pragma unroll
        for (int r = 0; r < 8; ++r) {
            float x = st[r];
            sum += x; ssq += x * x; mx = fmaxf(mx, x);
        }
    }

    sum += __shfl_xor(sum, 16, 32);
    ssq += __shfl_xor(ssq, 16, 32);
    mx  = fmaxf(mx, __shfl_xor(mx, 16, 32));

    if (lane < 16) {
        float mean = sum * (1.f / N_SEQ);
        float var  = ssq * (1.f / N_SEQ) - mean * mean;
        float c    = rsqrtf(var + 1e-5f) * 0.0625f;    // * EMBED_DIM^-0.5
        int row = b * N_SEQ + qblk + wave * 16 + l16;
        cRow[row] = c;
        cMax[row] = c * mx;
    }
}

// ---------------------------------------------------------------------------
// Pass 2: O = softmax(c * q k^T) v.  K tile via TDM (overlapped with the
// manual V transpose staging); S^T accumulators map in-lane onto the P
// A-fragment of the P.V WMMA.
// ---------------------------------------------------------------------------
__global__ void __launch_bounds__(256)
attn_out_kernel(const unsigned short* __restrict__ Qb,
                const unsigned short* __restrict__ Kb,
                const unsigned short* __restrict__ Vb,
                const float* __restrict__ cRow, const float* __restrict__ cMax,
                float* __restrict__ O)
{
    __shared__ unsigned short ldsK [32 * C_DIM];   // [m][c]  16 KB
    __shared__ unsigned short ldsVT[C_DIM * 32];   // [c][m]  16 KB (transposed)

    const int tid  = threadIdx.x;
    const int lane = tid & 31;
    const int wave = tid >> 5;
    const int half = (lane >> 4) & 1;
    const int l16  = lane & 15;
    const int b    = blockIdx.x >> 5;
    const int qblk = (blockIdx.x & 31) * 128;

    const int qrow = b * N_SEQ + qblk + wave * 16 + l16;
    const unsigned short* Qrow  = Qb + (size_t)qrow * C_DIM;
    const unsigned short* Kbase = Kb + (size_t)b * N_SEQ * C_DIM;
    const unsigned short* Vbase = Vb + (size_t)b * N_SEQ * C_DIM;
#if USE_TDM
    const unsigned ldsKoff = lds_addr_of(ldsK);
#endif

    const float c2 = cRow[qrow] * LOG2E;
    const float m2 = cMax[qrow] * LOG2E;

    Frag qf[8];
    #pragma unroll
    for (int s = 0; s < 8; ++s) {
        qf[s].q[0] = *(const uint4*)(Qrow + s * 32 + 8 * half);
        qf[s].q[1] = *(const uint4*)(Qrow + s * 32 + 16 + 8 * half);
    }

    v8f oacc[16] = {};
    float denom = 0.f;

    for (int mb = 0; mb < N_SEQ; mb += 32) {
        __syncthreads();
#if USE_TDM
        if (wave == 0) {                          // issue DMA, overlap V staging
            tdm_load_tile_2d(ldsKoff, Kbase + (size_t)mb * C_DIM,
                             C_DIM / 2, 32, N_SEQ - mb);
        }
#else
        #pragma unroll
        for (int i = 0; i < 4; ++i) {
            int idx = tid + i * 256;
            int r  = idx >> 5;
            int c8 = (idx & 31) * 8;
            *(uint4*)(&ldsK[r * C_DIM + c8]) =
                *(const uint4*)(Kbase + (size_t)(mb + r) * C_DIM + c8);
        }
#endif
        // V transpose staging: ldsVT[c][m]
        #pragma unroll
        for (int i = 0; i < 4; ++i) {
            int idx = tid + i * 256;
            int m  = idx >> 5;
            int c8 = (idx & 31) * 8;
            union { uint4 u; unsigned short s[8]; } t;
            t.u = *(const uint4*)(Vbase + (size_t)(mb + m) * C_DIM + c8);
            #pragma unroll
            for (int j = 0; j < 8; ++j) ldsVT[(c8 + j) * 32 + m] = t.s[j];
        }
        if (mb + 32 < N_SEQ)                      // global_prefetch_b8 next V tile
            __builtin_prefetch(Vbase + (size_t)(mb + 32) * C_DIM + tid * 32, 0, 3);
#if USE_TDM
        if (wave == 0) __builtin_amdgcn_s_wait_tensorcnt(0);
#endif
        __syncthreads();

        v8f s0 = {}, s1 = {};
        #pragma unroll
        for (int s = 0; s < 8; ++s) {
            Frag a0, a1;
            const unsigned short* k0 = &ldsK[l16 * C_DIM + s * 32];
            const unsigned short* k1 = &ldsK[(16 + l16) * C_DIM + s * 32];
            a0.q[0] = *(const uint4*)(k0 + 8 * half);
            a0.q[1] = *(const uint4*)(k0 + 16 + 8 * half);
            a1.q[0] = *(const uint4*)(k1 + 8 * half);
            a1.q[1] = *(const uint4*)(k1 + 16 + 8 * half);
            s0 = wmma_bf16(a0, qf[s], s0);
            s1 = wmma_bf16(a1, qf[s], s1);
        }

        Frag ap;   // element i<8 <- s0[i], i>=8 <- s1[i-8] (in-lane remap)
        #pragma unroll
        for (int r = 0; r < 8; ++r) {
            float p0 = __builtin_exp2f(c2 * s0[r] - m2);
            float p1 = __builtin_exp2f(c2 * s1[r] - m2);
            denom += p0 + p1;
            ap.v[r]     = __builtin_bit_cast(__bf16, f2bf(p0));
            ap.v[r + 8] = __builtin_bit_cast(__bf16, f2bf(p1));
        }

        #pragma unroll
        for (int ct = 0; ct < 16; ++ct) {
            Frag bv;
            const unsigned short* vr = &ldsVT[(ct * 16 + l16) * 32 + 8 * half];
            bv.q[0] = *(const uint4*)(vr);
            bv.q[1] = *(const uint4*)(vr + 16);
            oacc[ct] = wmma_bf16(ap, bv, oacc[ct]);
        }
    }

    denom += __shfl_xor(denom, 16, 32);

    #pragma unroll
    for (int ct = 0; ct < 16; ++ct) {
        #pragma unroll
        for (int r = 0; r < 8; ++r) {
            float d = __shfl(denom, r + 8 * half, 32);
            int qr = qblk + wave * 16 + r + 8 * half;
            O[((size_t)b * N_SEQ + qr) * C_DIM + ct * 16 + l16] = oacc[ct][r] / d;
        }
    }
}

// ---------------------------------------------------------------------------
extern "C" void kernel_launch(void* const* d_in, const int* in_sizes, int n_in,
                              void* d_out, int out_size, void* d_ws, size_t ws_size,
                              hipStream_t stream) {
    (void)in_sizes; (void)n_in; (void)out_size; (void)ws_size;

    const float* query = (const float*)d_in[0];
    const float* key   = (const float*)d_in[1];
    const float* value = (const float*)d_in[2];
    const float* Wq = (const float*)d_in[3];  const float* bq = (const float*)d_in[4];
    const float* Wk = (const float*)d_in[5];  const float* bk = (const float*)d_in[6];
    const float* Wv = (const float*)d_in[7];  const float* bv = (const float*)d_in[8];
    const float* Wm = (const float*)d_in[9];  const float* bm = (const float*)d_in[10];

    const int M = 4 * N_SEQ;                       // 16384 flattened rows
    char* ws = (char*)d_ws;
    const size_t bfBytes = (size_t)M * C_DIM * 2;  // 8 MB each
    unsigned short* qbf = (unsigned short*)(ws);
    unsigned short* kbf = (unsigned short*)(ws + bfBytes);
    unsigned short* vbf = (unsigned short*)(ws + 2 * bfBytes);
    float* Obuf = (float*)(ws + 3 * bfBytes);                        // 16 MB
    float* cRow = (float*)(ws + 3 * bfBytes + (size_t)M * C_DIM * 4);
    float* cMax = cRow + M;

    dim3 gemmGrid(M / 128, C_DIM / 64);
    dim3 attnGrid(4 * (N_SEQ / 128));
    dim3 blk(256);

    gemm256_kernel<<<gemmGrid, blk, 0, stream>>>(query, nullptr, Wq, bq, qbf, nullptr, M);
    gemm256_kernel<<<gemmGrid, blk, 0, stream>>>(key,   nullptr, Wk, bk, kbf, nullptr, M);
    gemm256_kernel<<<gemmGrid, blk, 0, stream>>>(value, nullptr, Wv, bv, vbf, nullptr, M);
    attn_stats_kernel<<<attnGrid, blk, 0, stream>>>(qbf, kbf, cRow, cMax);
    attn_out_kernel<<<attnGrid, blk, 0, stream>>>(qbf, kbf, vbf, cRow, cMax, Obuf);
    gemm256_kernel<<<gemmGrid, blk, 0, stream>>>(Obuf, query, Wm, bm, nullptr, (float*)d_out, M);
}
